// BlockWiseDistanceComputation_16870631539359
// MI455X (gfx1250) — matrix operand order repaired and checked by reading, not compiled, not run on hardware
//
#include <hip/hip_runtime.h>
#include <math.h>

// ---------------------------------------------------------------------------
// Hyperbolic (Poincare) block-wise attention for MI455X (gfx1250, wave32).
// All GEMMs (QKV proj, QK^T, E*V, out proj) use v_wmma_f32_16x16x32_bf16.
// ---------------------------------------------------------------------------

typedef __attribute__((ext_vector_type(8)))  __bf16 v8bf;
typedef __attribute__((ext_vector_type(16))) __bf16 v16bf;
typedef __attribute__((ext_vector_type(8)))  float  v8f;

#define NDIM 2048
#define DDIM 512
#define HH   8
#define DH   64
#define EPSF 1e-6f

union BF16Frag { v16bf v; v8bf h[2]; };

static __device__ inline v8f wmma_bf16(v16bf a, v16bf b, v8f c) {
    return __builtin_amdgcn_wmma_f32_16x16x32_bf16(
        /*neg_a=*/false, a, /*neg_b=*/false, b,
        /*c_mod=*/(short)0, c, /*reuse_a=*/false, /*reuse_b=*/false);
}

// ---------------------------------------------------------------------------
__global__ void cvt_bf16(const float* __restrict__ s, __bf16* __restrict__ d, int n) {
    int i = blockIdx.x * blockDim.x + threadIdx.x;
    if (i < n) d[i] = (__bf16)s[i];
}

// ---------------------------------------------------------------------------
// C[M x Nd] = A[M x Kd] (bf16) * B[Kd x Nd] (bf16) + bias
// 128x128 tile per workgroup, 8 waves arranged 4(m) x 2(n); each wave owns
// a 32x64 patch = 2x4 WMMA 16x16 accumulator tiles.
// F32OUT selects the output type at compile time (branch-free epilogue).
template <bool F32OUT>
__global__ __launch_bounds__(256) void gemm_bf16(
    const __bf16* __restrict__ A, const __bf16* __restrict__ B,
    const float* __restrict__ bias, float* __restrict__ Cf,
    __bf16* __restrict__ Cb, int M, int Kd, int Nd)
{
    __shared__ __bf16 sA [128][48];   // [m][k] tile, padded stride
    __shared__ __bf16 sBt[128][48];   // [n][k] tile (B transposed), padded

    const int t = threadIdx.x;
    const int lane = t & 31, wave = t >> 5;
    const int half = lane >> 4, l15 = lane & 15;
    const int wm = wave & 3, wn = wave >> 2;
    const int m0 = blockIdx.y * 128, n0 = blockIdx.x * 128;

    v8f acc[2][4];
    for (int i = 0; i < 2; i++) for (int j = 0; j < 4; j++) acc[i][j] = {};

    const int ar = t >> 1, ac = (t & 1) * 16;        // sA fill coords
    const int bkr = t & 31, bns = (t >> 5) * 16;     // sBt fill coords

    for (int k0 = 0; k0 < Kd; k0 += 32) {
        __syncthreads();
        {   // stage A tile (128x32)
            const __bf16* src = A + (size_t)(m0 + ar) * Kd + k0 + ac;
            *(v8bf*)&sA[ar][ac]     = *(const v8bf*)(src);
            *(v8bf*)&sA[ar][ac + 8] = *(const v8bf*)(src + 8);
        }
        {   // stage B tile transposed (32x128 -> [n][k])
            const __bf16* src = B + (size_t)(k0 + bkr) * Nd + n0 + bns;
            v8bf b0 = *(const v8bf*)(src), b1 = *(const v8bf*)(src + 8);
            #pragma unroll
            for (int j = 0; j < 8; j++) {
                sBt[bns + j][bkr]     = b0[j];
                sBt[bns + 8 + j][bkr] = b1[j];
            }
        }
        // prefetch next K-tile into cache while WMMAs run (global_prefetch_b8)
        if (k0 + 32 < Kd) {
            __builtin_prefetch(A + (size_t)(m0 + ar) * Kd + k0 + 32 + ac, 0, 0);
            __builtin_prefetch(B + (size_t)(k0 + 32 + bkr) * Nd + n0 + bns, 0, 0);
        }
        __syncthreads();

        BF16Frag afr[2]; v16bf bfr[4];
        #pragma unroll
        for (int mt = 0; mt < 2; mt++) {
            int ml = wm * 32 + mt * 16 + l15;
            afr[mt].h[0] = *(const v8bf*)&sA[ml][half * 8];
            afr[mt].h[1] = *(const v8bf*)&sA[ml][half * 8 + 16];
        }
        #pragma unroll
        for (int nt = 0; nt < 4; nt++) {
            int nl = wn * 64 + nt * 16 + l15;
            bfr[nt] = __builtin_shufflevector(*(const v8bf*)&sBt[nl][half * 8], *(const v8bf*)&sBt[nl][half * 8 + 16], 0,1,2,3,4,5,6,7,8,9,10,11,12,13,14,15);
        }
        #pragma unroll
        for (int mt = 0; mt < 2; mt++)
            #pragma unroll
            for (int nt = 0; nt < 4; nt++)
                acc[mt][nt] = wmma_bf16(afr[mt].v, bfr[nt], acc[mt][nt]);
    }

    // branch-free epilogue; bias hoisted per column tile
    #pragma unroll
    for (int nt = 0; nt < 4; nt++) {
        const int n = n0 + wn * 64 + nt * 16 + l15;
        const float bv = bias[n];
        #pragma unroll
        for (int mt = 0; mt < 2; mt++)
            #pragma unroll
            for (int r = 0; r < 8; r++) {
                int m = m0 + wm * 32 + mt * 16 + r + 8 * half;
                float v = acc[mt][nt][r] + bv;
                if constexpr (F32OUT) Cf[(size_t)m * Nd + n] = v;
                else                  Cb[(size_t)m * Nd + n] = (__bf16)v;
            }
    }
}

// ---------------------------------------------------------------------------
// per-(head,row) squared norms of bf16-rounded Q/K (consistent with WMMA qk)
__global__ void head_norms(const __bf16* __restrict__ Qb, const __bf16* __restrict__ Kb,
                           float* __restrict__ qn, float* __restrict__ kn)
{
    int idx = blockIdx.x * blockDim.x + threadIdx.x;   // 0..H*N-1
    int h = idx >> 11, n = idx & 2047;
    const __bf16* q = Qb + (size_t)n * DDIM + h * DH;
    const __bf16* k = Kb + (size_t)n * DDIM + h * DH;
    float sq = 0.f, sk = 0.f;
    #pragma unroll 8
    for (int d = 0; d < DH; d++) {
        float a = (float)q[d]; sq += a * a;
        float b = (float)k[d]; sk += b * b;
    }
    qn[h * NDIM + n] = sq;
    kn[h * NDIM + n] = sk;
}

// ---------------------------------------------------------------------------
// one workgroup = one (head, 128-row query block); loops causal key blocks.
__global__ __launch_bounds__(256) void hyp_attn(
    const __bf16* __restrict__ Qb, const __bf16* __restrict__ Kb,
    const __bf16* __restrict__ Vb, const float* __restrict__ qn,
    const float* __restrict__ kn, const float* __restrict__ cptr,
    __bf16* __restrict__ Ob)
{
    __shared__ __bf16 sVt[64][144];     // V block transposed: [d][key]
    __shared__ __bf16 sE [128][136];    // exp(scores) tile (bf16)
    __shared__ float  sqn[128], skn[128], srm[2][128], snorm[128];

    const int t = threadIdx.x, lane = t & 31, wave = t >> 5;
    const int half = lane >> 4, l15 = lane & 15;
    const int wm = wave & 3, wn = wave >> 2;   // scores: 32 rows x 64 keycols
                                               // output: 32 rows x 32 dcols
    const int h = blockIdx.y, qb = blockIdx.x;
    const int m0 = qb * 128;

    const float cc  = fmaxf(fabsf(cptr[0]), 1e-6f);
    const float rsc = 1.0f / sqrtf(cc);

    if (t < 128) { sqn[t] = qn[h * NDIM + m0 + t]; snorm[t] = 0.f; }

    v8f oacc[2][2];
    for (int a = 0; a < 2; a++) for (int b = 0; b < 2; b++) oacc[a][b] = {};

    const __bf16* Qh = Qb + h * DH;
    const __bf16* Kh = Kb + h * DH;
    const __bf16* Vh = Vb + h * DH;

    for (int kb = 0; kb <= qb; kb++) {
        __syncthreads();
        if (t < 128) skn[t] = kn[h * NDIM + kb * 128 + t];
        {   // stage V block transposed into LDS
            int key = t >> 1, ds = (t & 1) * 32;
            const __bf16* src = Vh + (size_t)(kb * 128 + key) * DDIM + ds;
            v8bf v0 = *(const v8bf*)(src),      v1 = *(const v8bf*)(src + 8);
            v8bf v2 = *(const v8bf*)(src + 16), v3 = *(const v8bf*)(src + 24);
            #pragma unroll
            for (int j = 0; j < 8; j++) {
                sVt[ds + j][key]      = v0[j];
                sVt[ds + 8 + j][key]  = v1[j];
                sVt[ds + 16 + j][key] = v2[j];
                sVt[ds + 24 + j][key] = v3[j];
            }
        }
        __syncthreads();

        // ---- S = Q * K^T (Dh=64 -> 2 WMMA k-steps) ----
        v8f s[2][4];
        for (int i = 0; i < 2; i++) for (int j = 0; j < 4; j++) s[i][j] = {};
        #pragma unroll
        for (int ks = 0; ks < 2; ks++) {
            BF16Frag afr[2]; v16bf bfr[4];
            #pragma unroll
            for (int mt = 0; mt < 2; mt++) {
                int ml = m0 + wm * 32 + mt * 16 + l15;
                const __bf16* src = Qh + (size_t)ml * DDIM + ks * 32 + half * 8;
                afr[mt].h[0] = *(const v8bf*)(src);
                afr[mt].h[1] = *(const v8bf*)(src + 16);
            }
            #pragma unroll
            for (int nt = 0; nt < 4; nt++) {
                int nl = kb * 128 + wn * 64 + nt * 16 + l15;
                bfr[nt] = __builtin_shufflevector(*(const v8bf*)(Kh + (size_t)nl * DDIM + ks * 32 + half * 8), *(const v8bf*)(Kh + (size_t)nl * DDIM + ks * 32 + half * 8 + 16), 0,1,2,3,4,5,6,7,8,9,10,11,12,13,14,15);
            }
            #pragma unroll
            for (int mt = 0; mt < 2; mt++)
                #pragma unroll
                for (int nt = 0; nt < 4; nt++)
                    s[mt][nt] = wmma_bf16(afr[mt].v, bfr[nt], s[mt][nt]);
        }

        // ---- Poincare distance scores + causal mask + block row-max ----
        const bool diag = (kb == qb);
        #pragma unroll
        for (int mt = 0; mt < 2; mt++) {
            #pragma unroll
            for (int r = 0; r < 8; r++) {
                int mloc = wm * 32 + mt * 16 + r + 8 * half;
                float qv = sqn[mloc];
                float rowm = -3.0e38f;
                #pragma unroll
                for (int nt = 0; nt < 4; nt++) {
                    int nloc = wn * 64 + nt * 16 + l15;
                    float kv = skn[nloc];
                    float qk = s[mt][nt][r];
                    float d2  = fmaxf(qv + kv - 2.f * qk, 0.f);
                    float den = fmaxf((1.f - cc * qv) * (1.f - cc * kv), EPSF);
                    float arg = fmaxf(1.f + 2.f * cc * d2 / den, 1.f + EPSF);
                    float v = -acoshf(arg) * rsc;
                    if (diag && nloc > mloc) v = -3.0e38f;
                    s[mt][nt][r] = v;
                    rowm = fmaxf(rowm, v);
                }
                rowm = fmaxf(rowm, __shfl_xor(rowm, 1, 32));
                rowm = fmaxf(rowm, __shfl_xor(rowm, 2, 32));
                rowm = fmaxf(rowm, __shfl_xor(rowm, 4, 32));
                rowm = fmaxf(rowm, __shfl_xor(rowm, 8, 32));
                if (l15 == 0) srm[wn][mloc] = rowm;
            }
        }
        __syncthreads();

        // ---- e = exp(s - blockmax), accumulate normalizer, stage E ----
        #pragma unroll
        for (int mt = 0; mt < 2; mt++) {
            #pragma unroll
            for (int r = 0; r < 8; r++) {
                int mloc = wm * 32 + mt * 16 + r + 8 * half;
                float rmax = fmaxf(srm[0][mloc], srm[1][mloc]);
                float rs = 0.f;
                #pragma unroll
                for (int nt = 0; nt < 4; nt++) {
                    int nloc = wn * 64 + nt * 16 + l15;
                    float v = s[mt][nt][r];
                    float e = (v < -1.0e38f) ? 0.f : __expf(v - rmax);
                    sE[mloc][nloc] = (__bf16)e;
                    rs += e;
                }
                rs += __shfl_xor(rs, 1, 32);
                rs += __shfl_xor(rs, 2, 32);
                rs += __shfl_xor(rs, 4, 32);
                rs += __shfl_xor(rs, 8, 32);
                if (l15 == 0) atomicAdd(&snorm[mloc], rs);
            }
        }
        __syncthreads();

        // ---- O += E * V (K=128 keys -> 4 WMMA k-steps) ----
        #pragma unroll
        for (int ks = 0; ks < 4; ks++) {
            BF16Frag afr[2]; v16bf bfr[2];
            #pragma unroll
            for (int mt = 0; mt < 2; mt++) {
                int ml = wm * 32 + mt * 16 + l15;
                afr[mt].h[0] = *(const v8bf*)&sE[ml][ks * 32 + half * 8];
                afr[mt].h[1] = *(const v8bf*)&sE[ml][ks * 32 + half * 8 + 16];
            }
            #pragma unroll
            for (int dt = 0; dt < 2; dt++) {
                int dl = wn * 32 + dt * 16 + l15;
                bfr[dt] = __builtin_shufflevector(*(const v8bf*)&sVt[dl][ks * 32 + half * 8], *(const v8bf*)&sVt[dl][ks * 32 + half * 8 + 16], 0,1,2,3,4,5,6,7,8,9,10,11,12,13,14,15);
            }
            #pragma unroll
            for (int mt = 0; mt < 2; mt++)
                #pragma unroll
                for (int dt = 0; dt < 2; dt++)
                    oacc[mt][dt] = wmma_bf16(afr[mt].v, bfr[dt], oacc[mt][dt]);
        }
    }
    __syncthreads();

    // ---- normalize and emit attention output (bf16, head-major columns) ----
    #pragma unroll
    for (int mt = 0; mt < 2; mt++)
        #pragma unroll
        for (int dt = 0; dt < 2; dt++)
            #pragma unroll
            for (int r = 0; r < 8; r++) {
                int mloc = wm * 32 + mt * 16 + r + 8 * half;
                int dloc = wn * 32 + dt * 16 + l15;
                float nrm = fmaxf(snorm[mloc], EPSF);
                Ob[(size_t)(m0 + mloc) * DDIM + h * DH + dloc] =
                    (__bf16)(oacc[mt][dt][r] / nrm);
            }
}

// ---------------------------------------------------------------------------
extern "C" void kernel_launch(void* const* d_in, const int* in_sizes, int n_in,
                              void* d_out, int out_size, void* d_ws, size_t ws_size,
                              hipStream_t stream)
{
    (void)in_sizes; (void)n_in; (void)out_size; (void)ws_size;
    const float* x  = (const float*)d_in[0];
    const float* c  = (const float*)d_in[1];
    const float* Wq = (const float*)d_in[2];
    const float* bq = (const float*)d_in[3];
    const float* Wk = (const float*)d_in[4];
    const float* bk = (const float*)d_in[5];
    const float* Wv = (const float*)d_in[6];
    const float* bv = (const float*)d_in[7];
    const float* Wo = (const float*)d_in[8];
    const float* bo = (const float*)d_in[9];
    float* out = (float*)d_out;

    char* ws = (char*)d_ws;
    size_t off = 0;
    auto carve = [&](size_t bytes) -> char* {
        char* p = ws + off;
        off += (bytes + 255) & ~(size_t)255;
        return p;
    };
    __bf16* xb   = (__bf16*)carve((size_t)NDIM * DDIM * 2);
    __bf16* Wqb  = (__bf16*)carve((size_t)DDIM * DDIM * 2);
    __bf16* Wkb  = (__bf16*)carve((size_t)DDIM * DDIM * 2);
    __bf16* Wvb  = (__bf16*)carve((size_t)DDIM * DDIM * 2);
    __bf16* Wob  = (__bf16*)carve((size_t)DDIM * DDIM * 2);
    __bf16* Qbuf = (__bf16*)carve((size_t)NDIM * DDIM * 2);
    __bf16* Kbuf = (__bf16*)carve((size_t)NDIM * DDIM * 2);
    __bf16* Vbuf = (__bf16*)carve((size_t)NDIM * DDIM * 2);
    __bf16* Abuf = (__bf16*)carve((size_t)NDIM * DDIM * 2);
    float*  qnb  = (float*)carve((size_t)HH * NDIM * 4);
    float*  knb  = (float*)carve((size_t)HH * NDIM * 4);

    const int nX = NDIM * DDIM, nW = DDIM * DDIM;
    cvt_bf16<<<(nX + 255) / 256, 256, 0, stream>>>(x,  xb,  nX);
    cvt_bf16<<<(nW + 255) / 256, 256, 0, stream>>>(Wq, Wqb, nW);
    cvt_bf16<<<(nW + 255) / 256, 256, 0, stream>>>(Wk, Wkb, nW);
    cvt_bf16<<<(nW + 255) / 256, 256, 0, stream>>>(Wv, Wvb, nW);
    cvt_bf16<<<(nW + 255) / 256, 256, 0, stream>>>(Wo, Wob, nW);

    dim3 gg(DDIM / 128, NDIM / 128);   // (4, 16)
    gemm_bf16<false><<<gg, 256, 0, stream>>>(xb, Wqb, bq, nullptr, Qbuf, NDIM, DDIM, DDIM);
    gemm_bf16<false><<<gg, 256, 0, stream>>>(xb, Wkb, bk, nullptr, Kbuf, NDIM, DDIM, DDIM);
    gemm_bf16<false><<<gg, 256, 0, stream>>>(xb, Wvb, bv, nullptr, Vbuf, NDIM, DDIM, DDIM);

    head_norms<<<(HH * NDIM) / 256, 256, 0, stream>>>(Qbuf, Kbuf, qnb, knb);

    hyp_attn<<<dim3(NDIM / 128, HH), 256, 0, stream>>>(Qbuf, Kbuf, Vbuf,
                                                       qnb, knb, c, Abuf);

    gemm_bf16<true><<<gg, 256, 0, stream>>>(Abuf, Wob, bo, out, nullptr, NDIM, DDIM, DDIM);
}
